// CrossView3DAttention_50105088475510
// MI455X (gfx1250) — matrix operand order repaired and checked by reading, not compiled, not run on hardware
//
#include <hip/hip_runtime.h>

// ---------------------------------------------------------------------------
// CDNA5 (gfx1250, wave32) implementation of CrossView3DAttention
// WMMA bf16 GEMMs + double-buffered TDM staging + LDS-shared flash attention
// ---------------------------------------------------------------------------

typedef __bf16 bf16_t;
typedef __bf16 v16bf __attribute__((ext_vector_type(16)));
typedef __bf16 v8bf  __attribute__((ext_vector_type(8)));
typedef float  v8f   __attribute__((ext_vector_type(8)));
typedef unsigned int u32x4 __attribute__((ext_vector_type(4)));
typedef int i32x4 __attribute__((ext_vector_type(4)));
typedef int i32x8 __attribute__((ext_vector_type(8)));

#define T_TOK 1536
#define C_DIM 1024
#define B_SZ  2
#define M_TOT (B_SZ * T_TOK)   // 3072 rows

__device__ __forceinline__ v8f wmma_bf16(v16bf a, v16bf b, v8f c) {
    return __builtin_amdgcn_wmma_f32_16x16x32_bf16(
        /*neg_a=*/false, a, /*neg_b=*/false, b,
        /*c_mod=*/(short)0, c, /*reuse_a=*/false, /*reuse_b=*/false);
}

__device__ __forceinline__ v8bf pack8(float4 f0, float4 f1) {
    v8bf r;
    r[0] = (bf16_t)f0.x; r[1] = (bf16_t)f0.y; r[2] = (bf16_t)f0.z; r[3] = (bf16_t)f0.w;
    r[4] = (bf16_t)f1.x; r[5] = (bf16_t)f1.y; r[6] = (bf16_t)f1.z; r[7] = (bf16_t)f1.w;
    return r;
}

__device__ __forceinline__ v16bf pack16(float4 f0, float4 f1, float4 f2, float4 f3) {
    v16bf r;
    r[0]  = (bf16_t)f0.x; r[1]  = (bf16_t)f0.y; r[2]  = (bf16_t)f0.z; r[3]  = (bf16_t)f0.w;
    r[4]  = (bf16_t)f1.x; r[5]  = (bf16_t)f1.y; r[6]  = (bf16_t)f1.z; r[7]  = (bf16_t)f1.w;
    r[8]  = (bf16_t)f2.x; r[9]  = (bf16_t)f2.y; r[10] = (bf16_t)f2.z; r[11] = (bf16_t)f2.w;
    r[12] = (bf16_t)f3.x; r[13] = (bf16_t)f3.y; r[14] = (bf16_t)f3.z; r[15] = (bf16_t)f3.w;
    return r;
}

__device__ __forceinline__ v16bf cat8(v8bf lo, v8bf hi) {
    v16bf r;
#pragma unroll
    for (int i = 0; i < 8; ++i) { r[i] = lo[i]; r[8 + i] = hi[i]; }
    return r;
}

// ---------------------------------------------------------------------------
// TDM: DMA a rows x 32-element f32 tile (row stride = stride_elems) into LDS.
// D# layout per CDNA5 ISA (2D tensor).
// This toolchain uses the 6-arg builtin: (g0, g1, g2, g3, g4, cpol).
// ---------------------------------------------------------------------------
__device__ __forceinline__ void tdm_load_tile32(const float* gaddr, unsigned int lds_off,
                                                int rows, int rowlen_elems,
                                                int tot_rows, int stride_elems) {
    const unsigned long long ga = (unsigned long long)(uintptr_t)gaddr;
    u32x4 g0;
    g0[0] = 1u;                                             // count=1, user mode
    g0[1] = lds_off;                                        // lds_addr (bytes)
    g0[2] = (unsigned int)ga;                               // global_addr[31:0]
    g0[3] = (unsigned int)((ga >> 32) & 0x01FFFFFFu) | 0x80000000u; // addr[56:32] | type=2
    i32x8 g1;
    g1[0] = 0x00020000;                                     // data_size=2 -> 4 bytes
    g1[1] = (int)(((unsigned)rowlen_elems & 0xFFFFu) << 16);        // tensor_dim0[15:0]
    g1[2] = (int)(((unsigned)rowlen_elems >> 16) |
                  (((unsigned)tot_rows & 0xFFFFu) << 16));          // td0[31:16] | td1[15:0]
    g1[3] = (int)(((unsigned)tot_rows >> 16) | (32u << 16));        // td1[31:16] | tile_dim0=32
    g1[4] = rows;                                           // tile_dim1 (tile_dim2 = 0)
    g1[5] = stride_elems;                                   // tensor_dim0_stride[31:0]
    g1[6] = 0;
    g1[7] = 0;
    const i32x4 z4 = {0, 0, 0, 0};
    const i32x8 z8 = {0, 0, 0, 0, 0, 0, 0, 0};
    __builtin_amdgcn_tensor_load_to_lds(g0, g1, z4, z4, z8, 0);
}

// ---------------------------------------------------------------------------
// Mask build: mask[b][q][k] = same_view || dist^2 > 100, diagonal fix
// ---------------------------------------------------------------------------
__device__ __forceinline__ float n2n(float v) {
    if (v != v) return 0.0f;
    if (v > 3.0e38f) return 100.0f;
    if (v < -3.0e38f) return -100.0f;
    return v;
}

__global__ __launch_bounds__(256)
void mask_kernel(const float* __restrict__ coords, unsigned char* __restrict__ mask) {
    const int bq = blockIdx.x;
    const int b = bq / T_TOK, q = bq % T_TOK;
    const int tid = threadIdx.x;
    const float* cq = coords + (size_t)(b * T_TOK + q) * 3;
    const float qx = n2n(cq[0]), qy = n2n(cq[1]), qz = n2n(cq[2]);
    const int vq = q >> 8;  // tokens-per-view = 256
    __shared__ int allm;
    if (tid == 0) allm = 1;
    __syncthreads();
    unsigned char* mr = mask + (size_t)b * T_TOK * T_TOK + (size_t)q * T_TOK;
    int loc = 1;
    for (int k = tid; k < T_TOK; k += 256) {
        const float* ck = coords + (size_t)(b * T_TOK + k) * 3;
        const float dx = qx - n2n(ck[0]);
        const float dy = qy - n2n(ck[1]);
        const float dz = qz - n2n(ck[2]);
        const float d2 = dx * dx + dy * dy + dz * dz;
        const int msk = ((k >> 8) == vq) | (d2 > 100.0f);
        mr[k] = (unsigned char)msk;
        loc &= msk;
    }
    if (!loc) atomicAnd(&allm, 0);
    __syncthreads();
    if (tid == 0 && allm) mr[q] = 0;  // unmask diagonal for fully-masked rows
}

// ---------------------------------------------------------------------------
// Copy kernel (x = pos_embed)
// ---------------------------------------------------------------------------
__global__ void copy_kernel(const float* __restrict__ s, float* __restrict__ d, int n) {
    for (int i = blockIdx.x * blockDim.x + threadIdx.x; i < n; i += gridDim.x * blockDim.x)
        d[i] = s[i];
}

// ---------------------------------------------------------------------------
// LayerNorm: one block per row of 1024
// ---------------------------------------------------------------------------
__global__ __launch_bounds__(256)
void ln_kernel(const float* __restrict__ x, const float* __restrict__ g,
               const float* __restrict__ bta, float* __restrict__ y) {
    const int row = blockIdx.x;
    const int tid = threadIdx.x;
    const float* xr = x + (size_t)row * C_DIM;
    const float4 v = *(const float4*)(xr + tid * 4);
    float s  = v.x + v.y + v.z + v.w;
    float s2 = v.x * v.x + v.y * v.y + v.z * v.z + v.w * v.w;
#pragma unroll
    for (int off = 1; off < 32; off <<= 1) {
        s  += __shfl_xor(s,  off, 32);
        s2 += __shfl_xor(s2, off, 32);
    }
    __shared__ float red[8][2];
    if ((tid & 31) == 0) { red[tid >> 5][0] = s; red[tid >> 5][1] = s2; }
    __syncthreads();
    float S = 0.0f, S2 = 0.0f;
#pragma unroll
    for (int i = 0; i < 8; ++i) { S += red[i][0]; S2 += red[i][1]; }
    const float mean = S * (1.0f / C_DIM);
    const float var  = S2 * (1.0f / C_DIM) - mean * mean;
    const float rstd = rsqrtf(var + 1e-5f);
    const float4 gv = *(const float4*)(g + tid * 4);
    const float4 bv = *(const float4*)(bta + tid * 4);
    float4 o;
    o.x = (v.x - mean) * rstd * gv.x + bv.x;
    o.y = (v.y - mean) * rstd * gv.y + bv.y;
    o.z = (v.z - mean) * rstd * gv.z + bv.z;
    o.w = (v.w - mean) * rstd * gv.w + bv.w;
    *(float4*)(y + (size_t)row * C_DIM + tid * 4) = o;
}

// ---------------------------------------------------------------------------
// WMMA GEMM with double-buffered TDM staging:
//   out[m][n] = epi( A·W^T + bias [+res] )
// Block tile 64(M) x 128(N), BK=32. 8 waves, each owns a 16x64 strip.
// Wave 0 issues next k-step's TDM loads before computing the current one;
// s_wait_tensorcnt(2) retires the older pair (TDM is in-order per wave).
// EPI: 0 = store, 1 = +residual, 2 = exact GELU.
// ---------------------------------------------------------------------------
template <int EPI>
__global__ __launch_bounds__(256)
void gemm_kernel(const float* __restrict__ A, const float* __restrict__ W,
                 const float* __restrict__ bias, const float* __restrict__ res,
                 float* __restrict__ out, int M, int N, int K) {
    __shared__ __align__(16) float Asf[2][64][32];    // 16 KB
    __shared__ __align__(16) float Bsf[2][128][32];   // 32 KB

    const int tid  = threadIdx.x;
    const int lane = tid & 31;
    const int w    = tid >> 5;
    const int m0   = blockIdx.y * 64;
    const int n0   = blockIdx.x * 128;
    const int mi   = (w & 3) * 16;   // wave's M sub-tile
    const int nh   = (w >> 2) * 64;  // wave's N half

    const int ln16 = lane & 15;
    const int kba  = (lane < 16) ? 0 : 8;   // A-fragment k base
    const int kbb  = (lane < 16) ? 0 : 16;  // unused; B fragments also use kba

    const unsigned int lds_a0 = (unsigned int)(uintptr_t)&Asf[0][0][0];
    const unsigned int lds_a1 = (unsigned int)(uintptr_t)&Asf[1][0][0];
    const unsigned int lds_b0 = (unsigned int)(uintptr_t)&Bsf[0][0][0];
    const unsigned int lds_b1 = (unsigned int)(uintptr_t)&Bsf[1][0][0];

    const float* Abase = A + (size_t)m0 * K;
    const float* Wbase = W + (size_t)n0 * K;

    v8f acc[4] = {};

    // prologue: fill buffer 0
    if (w == 0) {
        tdm_load_tile32(Abase, lds_a0, 64,  K, M, K);
        tdm_load_tile32(Wbase, lds_b0, 128, K, N, K);
    }

    int cur = 0;
    for (int kb = 0; kb < K; kb += 32) {
        if (w == 0) {
            if (kb + 32 < K) {
                // issue next tiles into the other buffer, then retire current pair
                const unsigned int na = cur ? lds_a0 : lds_a1;
                const unsigned int nb = cur ? lds_b0 : lds_b1;
                tdm_load_tile32(Abase + kb + 32, na, 64,  K, M, K);
                tdm_load_tile32(Wbase + kb + 32, nb, 128, K, N, K);
                __builtin_amdgcn_s_wait_tensorcnt(2);
            } else {
                __builtin_amdgcn_s_wait_tensorcnt(0);
            }
        }
        __syncthreads();

        // A fragment: lane holds row (mi+ln16); two 8-float runs at kba, kba+16
        const float* arp = &Asf[cur][mi + ln16][0];
        const v16bf af = pack16(*(const float4*)(arp + kba),
                                *(const float4*)(arp + kba + 4),
                                *(const float4*)(arp + kba + 16),
                                *(const float4*)(arp + kba + 20));
#pragma unroll
        for (int ns = 0; ns < 4; ++ns) {
            const float* brp = &Bsf[cur][nh + ns * 16 + ln16][kba];
            const v16bf bf = pack16(*(const float4*)brp,
                                    *(const float4*)(brp + 4),
                                    *(const float4*)(brp + 16),
                                    *(const float4*)(brp + 20));
            acc[ns] = wmma_bf16(af, bf, acc[ns]);
        }
        __syncthreads();   // everyone done reading buf[cur] before it is refilled
        cur ^= 1;
    }

    // epilogue
    const int hi8 = (lane >= 16) ? 8 : 0;
#pragma unroll
    for (int ns = 0; ns < 4; ++ns) {
        const int n = n0 + nh + ns * 16 + ln16;
        const float bv = bias[n];
#pragma unroll
        for (int v = 0; v < 8; ++v) {
            const int m = m0 + mi + v + hi8;
            float val = acc[ns][v] + bv;
            if (EPI == 1) val += res[(size_t)m * N + n];
            if (EPI == 2) val = 0.5f * val * (1.0f + erff(val * 0.70710678118654752f));
            out[(size_t)m * N + n] = val;
        }
    }
}

// ---------------------------------------------------------------------------
// Flash attention: block = 8 waves sharing one (b,h); each wave owns a
// 16-query tile; K/V tiles (32 keys) cooperatively staged in LDS as bf16
// (V transposed so V B-fragments are contiguous LDS reads).
// ---------------------------------------------------------------------------
__global__ __launch_bounds__(256)
void attn_kernel(const float* __restrict__ qkv, const unsigned char* __restrict__ mask,
                 float* __restrict__ o) {
    __shared__ __align__(16) bf16_t Ks[32][64];     // 4 KB: K tile, row = key
    __shared__ __align__(16) bf16_t Vt[64][32];     // 4 KB: V tile transposed, row = d
    __shared__ __align__(16) bf16_t Pl[8][16][32];  // 8 KB: wave-private P tiles

    const int tid  = threadIdx.x;
    const int lane = tid & 31;
    const int w    = tid >> 5;
    const int bh   = blockIdx.x / 12;      // 12 query-groups per (b,h)
    const int mg   = blockIdx.x % 12;
    const int b    = bh >> 4;
    const int h    = bh & 15;
    const int m0   = (mg * 8 + w) * 16;    // this wave's 16-query tile
    const int hi8  = (lane >= 16) ? 8 : 0;
    const int ln16 = lane & 15;
    const int kba  = (lane < 16) ? 0 : 8;
    const int kbb  = (lane < 16) ? 0 : 16;
    const float scale = 0.125f;  // 1/sqrt(64)

    const int skey = tid >> 3;        // staging: key row 0..31
    const int sdg  = (tid & 7) * 8;   // staging: 8 head-dims

    // Q fragments (hd split into two 32-wide k-steps), from global once
    v16bf qa[2];
    {
        const float* qp0 = qkv + (size_t)(b * T_TOK + m0 + ln16) * 3072 + h * 64;
#pragma unroll
        for (int s = 0; s < 2; ++s) {
            const float* qp = qp0 + s * 32 + kba;
            qa[s] = pack16(*(const float4*)qp, *(const float4*)(qp + 4),
                           *(const float4*)(qp + 16), *(const float4*)(qp + 20));
        }
    }

    v8f oacc[4] = {};
    float mrow[8], lrow[8];
#pragma unroll
    for (int v = 0; v < 8; ++v) { mrow[v] = -1e30f; lrow[v] = 0.0f; }

    const unsigned char* mbase = mask + (size_t)b * T_TOK * T_TOK;

    for (int kt = 0; kt < 48; ++kt) {
        const int k0 = kt * 32;

        // cooperative stage of K (row-major) and V (transposed) as bf16
        {
            const size_t rowb = (size_t)(b * T_TOK + k0 + skey) * 3072 + h * 64 + sdg;
            const float* kgp = qkv + rowb + C_DIM;
            *(v8bf*)&Ks[skey][sdg] = pack8(*(const float4*)kgp, *(const float4*)(kgp + 4));
            const float* vgp = qkv + rowb + 2 * C_DIM;
            const float4 v0 = *(const float4*)vgp;
            const float4 v1 = *(const float4*)(vgp + 4);
            Vt[sdg + 0][skey] = (bf16_t)v0.x; Vt[sdg + 1][skey] = (bf16_t)v0.y;
            Vt[sdg + 2][skey] = (bf16_t)v0.z; Vt[sdg + 3][skey] = (bf16_t)v0.w;
            Vt[sdg + 4][skey] = (bf16_t)v1.x; Vt[sdg + 5][skey] = (bf16_t)v1.y;
            Vt[sdg + 6][skey] = (bf16_t)v1.z; Vt[sdg + 7][skey] = (bf16_t)v1.w;
        }
        __syncthreads();

        // S = Q * K^T for two 16-key tiles
        v8f sc[2] = {};
#pragma unroll
        for (int t = 0; t < 2; ++t) {
#pragma unroll
            for (int ss = 0; ss < 2; ++ss) {
                const bf16_t* kp = &Ks[t * 16 + ln16][ss * 32 + kba];
                sc[t] = wmma_bf16(qa[ss], cat8(*(const v8bf*)kp, *(const v8bf*)(kp + 16)),
                                  sc[t]);
            }
        }

        // mask + scale + online softmax stats (rows live in VGPR index)
        float p0[8], p1[8];
#pragma unroll
        for (int v = 0; v < 8; ++v) {
            const int m = m0 + v + hi8;
            const unsigned char* mrp = mbase + (size_t)m * T_TOK + k0;
            const float s0v = mrp[ln16]      ? -1e30f : sc[0][v] * scale;
            const float s1v = mrp[16 + ln16] ? -1e30f : sc[1][v] * scale;
            float mx = fmaxf(s0v, s1v);
            mx = fmaxf(mx, __shfl_xor(mx, 1, 32));
            mx = fmaxf(mx, __shfl_xor(mx, 2, 32));
            mx = fmaxf(mx, __shfl_xor(mx, 4, 32));
            mx = fmaxf(mx, __shfl_xor(mx, 8, 32));
            const float mn = fmaxf(mrow[v], mx);
            const float alpha = __expf(mrow[v] - mn);
            mrow[v] = mn;
            const float e0 = (s0v <= -1e29f) ? 0.0f : __expf(s0v - mn);
            const float e1 = (s1v <= -1e29f) ? 0.0f : __expf(s1v - mn);
            float rs = e0 + e1;
            rs += __shfl_xor(rs, 1, 32);
            rs += __shfl_xor(rs, 2, 32);
            rs += __shfl_xor(rs, 4, 32);
            rs += __shfl_xor(rs, 8, 32);
            lrow[v] = lrow[v] * alpha + rs;
#pragma unroll
            for (int dt = 0; dt < 4; ++dt) oacc[dt][v] *= alpha;
            p0[v] = e0; p1[v] = e1;
        }

        // round-trip P through LDS: C-layout -> A-fragment layout
#pragma unroll
        for (int v = 0; v < 8; ++v) {
            Pl[w][v + hi8][ln16]      = (bf16_t)p0[v];
            Pl[w][v + hi8][16 + ln16] = (bf16_t)p1[v];
        }
        __syncthreads();
        const bf16_t* prp = &Pl[w][ln16][0];
        const v16bf pf = cat8(*(const v8bf*)(prp + kba),
                              *(const v8bf*)(prp + kba + 16));

        // O += P * V   (V fragments: contiguous reads of transposed tile)
#pragma unroll
        for (int dt = 0; dt < 4; ++dt) {
            const bf16_t* vp = &Vt[dt * 16 + ln16][kba];
            oacc[dt] = wmma_bf16(pf, cat8(*(const v8bf*)vp, *(const v8bf*)(vp + 16)),
                                 oacc[dt]);
        }
        __syncthreads();  // all reads done before next staging overwrites
    }

    // epilogue: O /= rowsum, scatter to (B,T,C) with head concat
#pragma unroll
    for (int v = 0; v < 8; ++v) {
        const float inv = (lrow[v] > 0.0f) ? 1.0f / lrow[v] : 0.0f;
        const int m = m0 + v + hi8;
#pragma unroll
        for (int dt = 0; dt < 4; ++dt) {
            o[(size_t)(b * T_TOK + m) * C_DIM + h * 64 + dt * 16 + ln16] = oacc[dt][v] * inv;
        }
    }
}

// ---------------------------------------------------------------------------
// Host-side orchestration
// ---------------------------------------------------------------------------
extern "C" void kernel_launch(void* const* d_in, const int* in_sizes, int n_in,
                              void* d_out, int out_size, void* d_ws, size_t ws_size,
                              hipStream_t stream) {
    const float* pos    = (const float*)d_in[0];
    const float* coords = (const float*)d_in[1];
    const float* ln1g   = (const float*)d_in[2];
    const float* ln1b   = (const float*)d_in[3];
    const float* qkvw   = (const float*)d_in[4];
    const float* qkvb   = (const float*)d_in[5];
    const float* outw   = (const float*)d_in[6];
    const float* outb   = (const float*)d_in[7];
    const float* ln2g   = (const float*)d_in[8];
    const float* ln2b   = (const float*)d_in[9];
    const float* fw1    = (const float*)d_in[10];
    const float* fb1    = (const float*)d_in[11];
    const float* fw2    = (const float*)d_in[12];
    const float* fb2    = (const float*)d_in[13];
    const float* ng     = (const float*)d_in[14];
    const float* nb     = (const float*)d_in[15];

    char* ws = (char*)d_ws;
    float* x    = (float*)ws; ws += (size_t)M_TOT * C_DIM * 4;       // 12.6 MB
    float* xn   = (float*)ws; ws += (size_t)M_TOT * C_DIM * 4;       // 12.6 MB
    float* qkv  = (float*)ws; ws += (size_t)M_TOT * 3 * C_DIM * 4;   // 37.7 MB
    float* att  = (float*)ws; ws += (size_t)M_TOT * C_DIM * 4;       // 12.6 MB
    float* hbuf = (float*)ws; ws += (size_t)M_TOT * 4 * C_DIM * 4;   // 50.3 MB
    unsigned char* mask = (unsigned char*)ws;                        //  4.7 MB

    mask_kernel<<<B_SZ * T_TOK, 256, 0, stream>>>(coords, mask);
    copy_kernel<<<3072, 256, 0, stream>>>(pos, x, M_TOT * C_DIM);

    for (int l = 0; l < 2; ++l) {
        ln_kernel<<<M_TOT, 256, 0, stream>>>(x, ln1g + l * C_DIM, ln1b + l * C_DIM, xn);
        gemm_kernel<0><<<dim3(24, 48), 256, 0, stream>>>(
            xn, qkvw + (size_t)l * 3 * C_DIM * C_DIM, qkvb + (size_t)l * 3 * C_DIM,
            nullptr, qkv, M_TOT, 3 * C_DIM, C_DIM);
        attn_kernel<<<384, 256, 0, stream>>>(qkv, mask, att);
        gemm_kernel<1><<<dim3(8, 48), 256, 0, stream>>>(
            att, outw + (size_t)l * C_DIM * C_DIM, outb + (size_t)l * C_DIM,
            x, x, M_TOT, C_DIM, C_DIM);
        ln_kernel<<<M_TOT, 256, 0, stream>>>(x, ln2g + l * C_DIM, ln2b + l * C_DIM, xn);
        gemm_kernel<2><<<dim3(32, 48), 256, 0, stream>>>(
            xn, fw1 + (size_t)l * 4 * C_DIM * C_DIM, fb1 + (size_t)l * 4 * C_DIM,
            nullptr, hbuf, M_TOT, 4 * C_DIM, C_DIM);
        gemm_kernel<1><<<dim3(8, 48), 256, 0, stream>>>(
            hbuf, fw2 + (size_t)l * C_DIM * 4 * C_DIM, fb2 + (size_t)l * C_DIM,
            x, x, M_TOT, C_DIM, 4 * C_DIM);
    }
    ln_kernel<<<M_TOT, 256, 0, stream>>>(x, ng, nb, (float*)d_out);
}
